// ProtT5Conv1dCRF4AAClassifier_49770081026595
// MI455X (gfx1250) — matrix operand order repaired and checked by reading, not compiled, not run on hardware
//
#include <hip/hip_runtime.h>

// Problem constants (from reference): B=32, T=1024, D=1024, K=8
#define B_ 32
#define T_ 1024
#define D_ 1024
#define K_ 8
#define R_ (3 * D_)        // GEMM reduction length = 3072
#define NCHUNK (R_ / 32)   // 96 chunks of K=32 for wmma_f32_16x16x32_bf16

typedef __attribute__((ext_vector_type(16))) __bf16 v16bf;
typedef __attribute__((ext_vector_type(8)))  float  v8f;

// workspace layout
#define EM_ELEMS  (B_ * T_ * K_)          // 262144 floats (1 MB)
#define WF_ELEMS  (NCHUNK * 2 * 32 * 16)  // 98304 bf16 (192 KB): [chunk][hi/lo][lane][16]
#define EM_BYTES  (EM_ELEMS * 4)
#define WF_BYTES  (WF_ELEMS * 2)

// ---------------------------------------------------------------------------
// Kernel 0: pack conv_w (K,D,3) fp32 into per-chunk bf16 B-fragments (hi+lo),
// laid out exactly as the dense 16-bit B 32x16 WMMA operand expects:
//   lane l: n = l&15, half = l>>4 ; element e: K = (e<8 ? e : e+8) + 8*half.
// Also zeroes the per-batch score accumulators (used by score_kernel).
// ---------------------------------------------------------------------------
__global__ __launch_bounds__(256) void prep_kernel(const float* __restrict__ w,
                                                   __bf16* __restrict__ wfrag,
                                                   float* __restrict__ score) {
    int idx = blockIdx.x * 256 + threadIdx.x;
    if (idx < B_) score[idx] = 0.0f;
    if (idx >= NCHUNK * 32 * 16) return;
    int e    = idx & 15;
    int lane = (idx >> 4) & 31;
    int c    = idx >> 9;
    int n    = lane & 15;
    int half = lane >> 4;
    int kk   = c * 32 + (e < 8 ? half * 8 + e : 16 + half * 8 + (e - 8));   // reduction index r = j*D + d
    int j    = kk >> 10;                 // 0..2
    int d    = kk & (D_ - 1);
    float x  = (n < K_) ? w[(n * D_ + d) * 3 + j] : 0.0f;
    __bf16 hi = (__bf16)x;
    __bf16 lo = (__bf16)(x - (float)hi);
    wfrag[((size_t)(c * 2 + 0) * 32 + lane) * 16 + e] = hi;
    wfrag[((size_t)(c * 2 + 1) * 32 + lane) * 16 + e] = lo;
}

// ---------------------------------------------------------------------------
// Kernel 1: conv-as-GEMM with bf16x3 WMMA (f32 accumulate).
// One wave per 16(t) x 16(k-padded) tile: 2048 waves total.
// A fragment per ISA 16-bit A 16x32 layout:
//   lane: m = lane&15, half = lane>>4 ; elem e: K = (e<8 ? e : e+8) + 8*half
// -> two contiguous groups of 8 floats per lane (32B-aligned float4 loads).
// ---------------------------------------------------------------------------
__global__ __launch_bounds__(256) void conv_wmma_kernel(const float* __restrict__ emb,
                                                        const __bf16* __restrict__ wfrag,
                                                        const float* __restrict__ bias,
                                                        float* __restrict__ em) {
    int tid  = blockIdx.x * 256 + threadIdx.x;
    int lane = tid & 31;
    int wave = tid >> 5;               // 0..2047
    int b    = wave >> 6;              // batch
    int t0   = (wave & 63) << 4;       // tile row base in time
    int m    = lane & 15;
    int half = lane >> 4;
    int t    = t0 + m;

    v8f acc = {};
    for (int c = 0; c < NCHUNK; ++c) {
        int j     = c >> 5;             // 1024/32 = 32 chunks per tap
        int dbase = (c & 31) << 5;
        int tj    = t + j - 1;          // SAME padding, zero outside [0,T)
        bool valid = (tj >= 0) && (tj < T_);

        float4 f0, f1, f2, f3;
        if (valid) {
            const float4* p = (const float4*)(emb + ((size_t)b * T_ + tj) * D_
                                              + dbase + half * 8);
            f0 = p[0]; f1 = p[1];       // K offsets 0..7  (+8*half)
            f2 = p[4]; f3 = p[5];       // K offsets 16..23 (+8*half)
        } else {
            f0 = make_float4(0.f, 0.f, 0.f, 0.f);
            f1 = f0; f2 = f0; f3 = f0;
        }
        float xv[16] = { f0.x, f0.y, f0.z, f0.w,  f1.x, f1.y, f1.z, f1.w,
                         f2.x, f2.y, f2.z, f2.w,  f3.x, f3.y, f3.z, f3.w };
        v16bf ah, al;
#pragma unroll
        for (int i = 0; i < 16; ++i) {
            __bf16 h = (__bf16)xv[i];
            ah[i] = h;
            al[i] = (__bf16)(xv[i] - (float)h);
        }
        const v16bf* wp = (const v16bf*)(wfrag + ((size_t)c * 64 + lane) * 16);
        v16bf bh = wp[0];
        v16bf bl = wp[32];              // lo fragment: +32*16 bf16 elements

        // (Ah+Al)(Wh+Wl) ~= Ah*Wh + Ah*Wl + Al*Wh  (fp32-class accuracy)
        acc = __builtin_amdgcn_wmma_f32_16x16x32_bf16(false, ah, false, bh,
                                                      (short)0, acc, false, false);
        acc = __builtin_amdgcn_wmma_f32_16x16x32_bf16(false, ah, false, bl,
                                                      (short)0, acc, false, false);
        acc = __builtin_amdgcn_wmma_f32_16x16x32_bf16(false, al, false, bh,
                                                      (short)0, acc, false, false);
    }

    // C/D layout: lane n = lane&15; VGPR r holds row m = r + 8*(lane>>4)
    int n = lane & 15;
    if (n < K_) {
        float bv = bias[n];
#pragma unroll
        for (int r = 0; r < 8; ++r) {
            int mr = r + half * 8;
            em[((size_t)b * T_ + t0 + mr) * K_ + n] = acc[r] + bv;
        }
    }
}

// ---------------------------------------------------------------------------
// Kernel 2: numerator path score, one thread per (b,t), atomicAdd per batch.
// ---------------------------------------------------------------------------
__global__ __launch_bounds__(256) void score_kernel(const float* __restrict__ em,
                                                    const int* __restrict__ labels,
                                                    const int* __restrict__ mask,
                                                    const float* __restrict__ start,
                                                    const float* __restrict__ endt,
                                                    const float* __restrict__ trans,
                                                    float* __restrict__ score) {
    int idx = blockIdx.x * 256 + threadIdx.x;   // 0 .. B*T-1
    int b  = idx >> 10;
    int t  = idx & (T_ - 1);
    int bt = b * T_ + t;
    int tag = labels[bt];
    int mk  = mask[bt];
    float contrib = 0.0f;
    if (t == 0) {
        contrib = start[tag] + em[(size_t)bt * K_ + tag];
    } else if (mk) {
        int ptag = labels[bt - 1];
        contrib = trans[ptag * K_ + tag] + em[(size_t)bt * K_ + tag];
    }
    // last valid position: mask is a contiguous prefix
    if (mk && (t == T_ - 1 || mask[bt + 1] == 0)) contrib += endt[tag];
    atomicAdd(&score[b], contrib);
}

// ---------------------------------------------------------------------------
// Kernel 3: CRF forward recurrence (latency-bound). One block, 256 threads:
// 8 lanes per batch (state k'), 32 batches. Cross-state gather via intra-wave
// shuffles inside each 8-lane group; masked alpha update; final logsumexp and
// block reduction to the scalar loss.
// ---------------------------------------------------------------------------
__global__ __launch_bounds__(256) void crf_forward_kernel(const float* __restrict__ em,
                                                          const int* __restrict__ mask,
                                                          const float* __restrict__ start,
                                                          const float* __restrict__ endt,
                                                          const float* __restrict__ trans,
                                                          const float* __restrict__ score,
                                                          float* __restrict__ out) {
    int tidx  = threadIdx.x;
    int b     = tidx >> 3;          // batch 0..31
    int kp    = tidx & 7;           // destination state
    int base8 = (tidx & 31) & ~7;   // lane-group base within wave

    float tc[8];                    // trans[:, kp] column in registers
#pragma unroll
    for (int k = 0; k < 8; ++k) tc[k] = trans[k * K_ + kp];

    float alpha = start[kp] + em[(size_t)b * T_ * K_ + kp];

    for (int t = 1; t < T_; ++t) {
        float e = em[((size_t)b * T_ + t) * K_ + kp];
        float v[8];
#pragma unroll
        for (int k = 0; k < 8; ++k)
            v[k] = __shfl(alpha, base8 + k, 32) + tc[k];
        float mx = v[0];
#pragma unroll
        for (int k = 1; k < 8; ++k) mx = fmaxf(mx, v[k]);
        float s = 0.0f;
#pragma unroll
        for (int k = 0; k < 8; ++k) s += __expf(v[k] - mx);
        float nxt = mx + __logf(s) + e;
        alpha = mask[b * T_ + t] ? nxt : alpha;
    }

    // log_z = logsumexp over 8 lanes of (alpha + end[kp])
    float v  = alpha + endt[kp];
    float mx = v;
    mx = fmaxf(mx, __shfl_xor(mx, 1, 32));
    mx = fmaxf(mx, __shfl_xor(mx, 2, 32));
    mx = fmaxf(mx, __shfl_xor(mx, 4, 32));
    float s = __expf(v - mx);
    s += __shfl_xor(s, 1, 32);
    s += __shfl_xor(s, 2, 32);
    s += __shfl_xor(s, 4, 32);
    float logz = mx + __logf(s);

    __shared__ float part[B_];
    if (kp == 0) part[b] = logz - score[b];
    __syncthreads();
    if (tidx == 0) {
        float tot = 0.0f;
        for (int i = 0; i < B_; ++i) tot += part[i];
        out[0] = tot;
    }
}

// ---------------------------------------------------------------------------
extern "C" void kernel_launch(void* const* d_in, const int* in_sizes, int n_in,
                              void* d_out, int out_size, void* d_ws, size_t ws_size,
                              hipStream_t stream) {
    const float* emb    = (const float*)d_in[0];
    const float* conv_w = (const float*)d_in[1];
    const float* conv_b = (const float*)d_in[2];
    const float* start  = (const float*)d_in[3];
    const float* endt   = (const float*)d_in[4];
    const float* trans  = (const float*)d_in[5];
    const int*   mask   = (const int*)d_in[6];
    const int*   labels = (const int*)d_in[7];

    float*  em    = (float*)d_ws;
    __bf16* wfrag = (__bf16*)((char*)d_ws + EM_BYTES);
    float*  score = (float*)((char*)d_ws + EM_BYTES + WF_BYTES);
    float*  out   = (float*)d_out;

    // pack weights + zero score accumulators (49152 threads = 192 blocks)
    prep_kernel<<<(NCHUNK * 32 * 16) / 256, 256, 0, stream>>>(conv_w, wfrag, score);
    // conv GEMM: 2048 waves = 65536 threads = 256 blocks
    conv_wmma_kernel<<<(B_ * (T_ / 16) * 32) / 256, 256, 0, stream>>>(emb, wfrag, conv_b, em);
    // numerator score: B*T threads
    score_kernel<<<(B_ * T_) / 256, 256, 0, stream>>>(em, labels, mask, start, endt, trans, score);
    // sequential CRF forward + final reduction
    crf_forward_kernel<<<1, 256, 0, stream>>>(em, mask, start, endt, trans, score, out);
}
